// GNNEncoder_51677046506020
// MI455X (gfx1250) — matrix operand order repaired and checked by reading, not compiled, not run on hardware
//
#include <hip/hip_runtime.h>
#include <hip/hip_bf16.h>

// ---------------------------------------------------------------------------
// GNN encoder for MI455X (gfx1250, wave32).
// All 64-wide GEMMs use v_wmma_f32_16x16x32_f16 (16x16 tiles, K=32 chunks).
// Weights staged in LDS *transposed* (Wt[col][k], padded stride) so B
// fragments are two contiguous 16B chunks per lane -> ds_load_b128 x2.
// All tail handling via row/edge CLAMPING (no predicated loads -> no
// exec-mask branches, no per-element loadcnt drains). Guarded stores only.
// ---------------------------------------------------------------------------

#define EMB 64
#define LDK64 72     // padded K-stride (f16) for 64-deep transposed weights
#define LDK128 136   // padded K-stride for 128-deep transposed weights
#define TROW 72      // padded row stride for 16x64 A tiles
#define AROW 136     // padded row stride for 16x128 A tiles
#define HROW 72      // padded row stride for 16x64 h tiles

typedef __attribute__((ext_vector_type(16))) _Float16 v16h;
typedef __attribute__((ext_vector_type(8)))  float    v8f;

// --- WMMA fragment loaders (CDNA5 ISA 7.12.2 layouts, wave32) ---------------
// A (16x32 f16): row M = lane&15. lane<16: elems 0..7 -> K+0..7, 8..15 -> K+16..23
//                lane>=16: shift K by +8.  Two contiguous 16B chunks per lane.
__device__ __forceinline__ v16h load_a_frag(const _Float16* base, int ldm,
                                            int k0, int lane) {
  int m = lane & 15;
  int khalf = (lane >> 4) * 8;
  const _Float16* p = base + m * ldm + k0 + khalf;
  v16h a;
#pragma unroll
  for (int e = 0; e < 8; ++e) a[e] = p[e];
#pragma unroll
  for (int e = 0; e < 8; ++e) a[8 + e] = p[16 + e];
  return a;
}

// B (32x16 f16) from TRANSPOSED weights Wt[col][k] (stride ldk):
// col N = n0 + (lane&15); elems 0..7 -> K+8h..8h+7, 8..15 -> K+16+8h..+23+8h (h=lane>>4).
// Two contiguous 16B chunks per lane -> two ds_load_b128.
__device__ __forceinline__ v16h load_b_frag_t(const _Float16* Wt, int ldk,
                                              int k0, int n0, int lane) {
  int col = n0 + (lane & 15);
  int kb = k0 + ((lane >> 4) * 8);
  const _Float16* p = Wt + col * ldk + kb;
  v16h b;
#pragma unroll
  for (int e = 0; e < 8; ++e) { b[e] = p[e]; b[8 + e] = p[16 + e]; }
  return b;
}

#define WMMA_F16(a, b, c) \
  __builtin_amdgcn_wmma_f32_16x16x32_f16(false, (a), false, (b), (short)0, (c), false, false)

// --- zero workspace ---------------------------------------------------------
__global__ void zero_kernel(float* __restrict__ p, int n) {
  int i = blockIdx.x * 256 + threadIdx.x;
  if (i < n) p[i] = 0.f;
}

// --- feature embedding: relu(relu((x+sh)*sc @ W1 + b1) @ W2 + b2) -----------
__global__ void embed_kernel(const float* __restrict__ x, int N, int F,
                             const float* __restrict__ sh, const float* __restrict__ sc,
                             const float* __restrict__ W1, const float* __restrict__ b1,
                             const float* __restrict__ W2, const float* __restrict__ b2,
                             float* __restrict__ out) {
  __shared__ float h1[4][EMB];
  int slot = threadIdx.x >> 6;
  int node = blockIdx.x * 4 + slot;
  int nodec = node < N ? node : N - 1;        // clamp, store-guarded below
  int o = threadIdx.x & 63;
  float acc = 0.f;
  for (int f = 0; f < F; ++f) {
    float v = (x[nodec * F + f] + sh[f]) * sc[f];
    acc += v * W1[f * EMB + o];
  }
  acc = fmaxf(acc + b1[o], 0.f);
  h1[slot][o] = acc;
  __syncthreads();
  float a2 = b2[o];
  for (int k = 0; k < EMB; ++k) a2 += h1[slot][k] * W2[k * EMB + o];
  if (node < N) out[node * EMB + o] = fmaxf(a2, 0.f);
}

// stage a [K][64] f32 weight into LDS transposed f16 Wt[col][k] (stride ldk)
__device__ __forceinline__ void stage_wt(const float* __restrict__ W,
                                         _Float16* Wt, int K, int ldk,
                                         int tid, int nthreads) {
  for (int t = tid; t < K * EMB; t += nthreads) {
    int k = t >> 6, n = t & 63;           // coalesced global read
    Wt[n * ldk + k] = (_Float16)W[t];     // strided LDS write (one-time)
  }
}

// --- node projection: dst_f16 = src_f32 @ W(64x64) (+ bias) via WMMA --------
__global__ void proj_kernel(const float* __restrict__ src, int N,
                            const float* __restrict__ W, const float* __restrict__ bias,
                            _Float16* __restrict__ dst) {
  __shared__ _Float16 Wt[EMB * LDK64];
  __shared__ float bl[EMB];
  stage_wt(W, Wt, EMB, LDK64, threadIdx.x, blockDim.x);
  if (threadIdx.x < EMB) bl[threadIdx.x] = bias ? bias[threadIdx.x] : 0.f;
  __syncthreads();

  int wave = threadIdx.x >> 5, lane = threadIdx.x & 31;
  int row0 = (blockIdx.x * 8 + wave) * 16;
  if (row0 >= N) return;

  int m = lane & 15;
  int khalf = (lane >> 4) * 8;
  int row = row0 + m;
  int rowc = row < N ? row : N - 1;           // clamp: no predicated loads
  const float* srow = src + (size_t)rowc * EMB;

  v8f acc[4] = {};
#pragma unroll
  for (int kc = 0; kc < 2; ++kc) {
    v16h a;
    int kb = kc * 32 + khalf;
#pragma unroll
    for (int e = 0; e < 8; ++e) a[e] = (_Float16)srow[kb + e];
#pragma unroll
    for (int e = 0; e < 8; ++e) a[8 + e] = (_Float16)srow[kb + 16 + e];
#pragma unroll
    for (int nt = 0; nt < 4; ++nt) {
      v16h b = load_b_frag_t(Wt, LDK64, kc * 32, nt * 16, lane);
      acc[nt] = WMMA_F16(a, b, acc[nt]);
    }
  }
  int radd = (lane >> 4) * 8;
#pragma unroll
  for (int nt = 0; nt < 4; ++nt) {
    int col = nt * 16 + (lane & 15);
    float bv = bl[col];
#pragma unroll
    for (int r = 0; r < 8; ++r) {
      int rr = row0 + r + radd;
      if (rr < N) dst[(size_t)rr * EMB + col] = (_Float16)(acc[nt][r] + bv);
    }
  }
}

// --- edge message + scatter:  agg[i] += relu((lp[i]+ef*We+rp[j])*s) @ Wfin + bfin
__global__ void edge_kernel(const _Float16* __restrict__ lp, const _Float16* __restrict__ rp,
                            const int* __restrict__ idx_i, const int* __restrict__ idx_j,
                            const float* __restrict__ ef_raw,
                            const float* __restrict__ e_shift, const float* __restrict__ e_scale,
                            const float* __restrict__ W_edge,
                            const float* __restrict__ s_fin_p,
                            const float* __restrict__ W_fin, const float* __restrict__ b_fin,
                            float* __restrict__ agg, int E_) {
  __shared__ _Float16 Wt[EMB * LDK64];          // ~9.2 KB transposed W_fin
  __shared__ float bl[EMB];
  __shared__ float we[EMB];
  __shared__ _Float16 tmp[8][16 * TROW];        // ~18.4 KB
  __shared__ int iidx[8][16];

  stage_wt(W_fin, Wt, EMB, LDK64, threadIdx.x, blockDim.x);
  if (threadIdx.x < EMB) { bl[threadIdx.x] = b_fin[threadIdx.x]; we[threadIdx.x] = W_edge[threadIdx.x]; }
  __syncthreads();

  float esh = e_shift[0], esc = e_scale[0], sfin = s_fin_p[0];

  int wave = threadIdx.x >> 5, lane = threadIdx.x & 31;
  int e0 = (blockIdx.x * 8 + wave) * 16;

  // Phase A: lane pair (m, m+16) builds f16 row for edge m (fused relu).
  // Clamped edge id: invalid edges compute garbage rows but are never
  // scattered (iidx = -1), so no predicated loads and no zero-fill needed.
  int m = lane & 15, half = lane >> 4;
  int e = e0 + m;
  bool ev = e < E_;
  int ec = ev ? e : E_ - 1;
  int vi = idx_i[ec];
  int vj = idx_j[ec];
  float efv = (ef_raw[ec] + esh) * esc;
  if (half == 0) iidx[wave][m] = ev ? vi : -1;
  const _Float16* lpr = lp + (size_t)vi * EMB + half * 32;
  const _Float16* rpr = rp + (size_t)vj * EMB + half * 32;
  _Float16* trow = &tmp[wave][m * TROW + half * 32];
#pragma unroll
  for (int k = 0; k < 32; ++k) {
    float t = (float)lpr[k] + efv * we[half * 32 + k] + (float)rpr[k];
    trow[k] = (_Float16)fmaxf(t * sfin, 0.f);
  }
  // wave-local LDS produce->consume fence (producer lanes == consumer wave)
  asm volatile("s_wait_dscnt 0" ::: "memory");

  // Phase B: 16x64 @ 64x64 message GEMM
  v8f acc[4] = {};
#pragma unroll
  for (int kc = 0; kc < 2; ++kc) {
    v16h a = load_a_frag(&tmp[wave][0], TROW, kc * 32, lane);
#pragma unroll
    for (int nt = 0; nt < 4; ++nt) {
      v16h b = load_b_frag_t(Wt, LDK64, kc * 32, nt * 16, lane);
      acc[nt] = WMMA_F16(a, b, acc[nt]);
    }
  }
  // Phase C: scatter-add (segment_sum)
  int radd = (lane >> 4) * 8;
#pragma unroll
  for (int nt = 0; nt < 4; ++nt) {
    int col = nt * 16 + (lane & 15);
    float bv = bl[col];
#pragma unroll
    for (int r = 0; r < 8; ++r) {
      int target = iidx[wave][r + radd];
      if (target >= 0) atomicAdd(&agg[(size_t)target * EMB + col], acc[nt][r] + bv);
    }
  }
}

// --- output MLP: out = relu(concat(agg*s_post, right) @ Wo1 + bo1) @ Wo2 + bo2
__global__ void out_kernel(const float* __restrict__ agg, const float* __restrict__ right,
                           int N, const float* __restrict__ s_post_p,
                           const float* __restrict__ W_o1, const float* __restrict__ b_o1,
                           const float* __restrict__ W_o2, const float* __restrict__ b_o2,
                           float* __restrict__ dst) {
  __shared__ _Float16 W1t[EMB * LDK128];     // ~17.4 KB (transposed 128x64)
  __shared__ _Float16 W2t[EMB * LDK64];      // ~9.2 KB
  __shared__ float b1l[EMB], b2l[EMB];
  __shared__ _Float16 atile[4][16 * AROW];   // ~17.4 KB (4 waves/block)
  __shared__ _Float16 htile[4][16 * HROW];   // ~9.2 KB

  // W_o1 is [128][64]; transpose to W1t[col][k], k in 0..127
  for (int t = threadIdx.x; t < 128 * EMB; t += blockDim.x) {
    int k = t >> 6, n = t & 63;
    W1t[n * LDK128 + k] = (_Float16)W_o1[t];
  }
  stage_wt(W_o2, W2t, EMB, LDK64, threadIdx.x, blockDim.x);
  if (threadIdx.x < EMB) { b1l[threadIdx.x] = b_o1[threadIdx.x]; b2l[threadIdx.x] = b_o2[threadIdx.x]; }
  __syncthreads();
  float spost = s_post_p[0];

  int wave = threadIdx.x >> 5, lane = threadIdx.x & 31;
  int row0 = (blockIdx.x * 4 + wave) * 16;

  int lm = lane & 15, half = lane >> 4;
  int row = row0 + lm;
  int rs = row < N ? row : N - 1;            // clamp: no predicated loads
  // half 0 stages k=0..63 (agg * s_post); half 1 stages k=64..127 (right)
  _Float16* arow = &atile[wave][lm * AROW + half * 64];
  const float* src0 = (half == 0) ? (agg + (size_t)rs * EMB) : (right + (size_t)rs * EMB);
  float mult = (half == 0) ? spost : 1.f;
#pragma unroll
  for (int k = 0; k < 64; ++k)
    arow[k] = (_Float16)(src0[k] * mult);
  asm volatile("s_wait_dscnt 0" ::: "memory");

  v8f acc[4] = {};
#pragma unroll
  for (int kc = 0; kc < 4; ++kc) {
    v16h a = load_a_frag(&atile[wave][0], AROW, kc * 32, lane);
#pragma unroll
    for (int nt = 0; nt < 4; ++nt) {
      v16h b = load_b_frag_t(W1t, LDK128, kc * 32, nt * 16, lane);
      acc[nt] = WMMA_F16(a, b, acc[nt]);
    }
  }
  // h = relu(acc + b1) -> LDS (f16, padded row-major)
#pragma unroll
  for (int nt = 0; nt < 4; ++nt) {
    int col = nt * 16 + lm;
#pragma unroll
    for (int r = 0; r < 8; ++r)
      htile[wave][(r + half * 8) * HROW + col] = (_Float16)fmaxf(acc[nt][r] + b1l[col], 0.f);
  }
  asm volatile("s_wait_dscnt 0" ::: "memory");

  v8f acc2[4] = {};
#pragma unroll
  for (int kc = 0; kc < 2; ++kc) {
    v16h a = load_a_frag(&htile[wave][0], HROW, kc * 32, lane);
#pragma unroll
    for (int nt = 0; nt < 4; ++nt) {
      v16h b = load_b_frag_t(W2t, LDK64, kc * 32, nt * 16, lane);
      acc2[nt] = WMMA_F16(a, b, acc2[nt]);
    }
  }
#pragma unroll
  for (int nt = 0; nt < 4; ++nt) {
    int col = nt * 16 + lm;
#pragma unroll
    for (int r = 0; r < 8; ++r) {
      int rr = row0 + r + half * 8;
      if (rr < N) dst[(size_t)rr * EMB + col] = acc2[nt][r] + b2l[col];
    }
  }
}

// ---------------------------------------------------------------------------
extern "C" void kernel_launch(void* const* d_in, const int* in_sizes, int n_in,
                              void* d_out, int out_size, void* d_ws, size_t ws_size,
                              hipStream_t stream) {
  (void)n_in; (void)out_size; (void)ws_size;
  const int CF = 5, NL = 4;
  const int N = in_sizes[0] / CF;
  const int E = in_sizes[2];

  // input map (depth-first, dict insertion order)
  const float* cons_x = (const float*)d_in[0];
  const int*   ei     = (const int*)d_in[1];     // [2][E]
  const float* ef_raw = (const float*)d_in[2];
  const float* var_x  = (const float*)d_in[3];
  // emb_params 4..17
  const float* e_shift = (const float*)d_in[16];
  const float* e_scale = (const float*)d_in[17];
  // conv param groups: vc base 18, cv base 30
  // order: W_left,b_left,W_edge,W_right,s_fin,W_fin,b_fin,s_post,W_o1,b_o1,W_o2,b_o2
  auto P = [&](int base, int off) { return (const float*)d_in[base + off]; };

  float* C = (float*)d_out;                 // [N][64]
  float* V = C + (size_t)N * EMB;           // [N][64]

  _Float16* lp = (_Float16*)d_ws;                            // N*64 f16
  _Float16* rp = lp + (size_t)N * EMB;                       // N*64 f16
  float*    agg = (float*)(rp + (size_t)N * EMB);            // N*64 f32

  const int* ei0 = ei;
  const int* ei1 = ei + E;

  // --- embeddings ---
  dim3 eb(256);
  embed_kernel<<<(N + 3) / 4, eb, 0, stream>>>(cons_x, N, 5,
      (const float*)d_in[4], (const float*)d_in[5], (const float*)d_in[6],
      (const float*)d_in[7], (const float*)d_in[8], (const float*)d_in[9], C);
  embed_kernel<<<(N + 3) / 4, eb, 0, stream>>>(var_x, N, 19,
      (const float*)d_in[10], (const float*)d_in[11], (const float*)d_in[12],
      (const float*)d_in[13], (const float*)d_in[14], (const float*)d_in[15], V);

  const int node_tiles  = (N + 15) / 16;
  const int proj_blocks = (node_tiles + 7) / 8;
  const int out_blocks  = (node_tiles + 3) / 4;
  const int edge_tiles  = (E + 15) / 16;
  const int edge_blocks = (edge_tiles + 7) / 8;
  const int zero_blocks = (N * EMB + 255) / 256;

  auto conv = [&](const float* left, const float* right, float* dst,
                  const int* irow, const int* jrow, int base, int l) {
    proj_kernel<<<proj_blocks, 256, 0, stream>>>(right, N,
        P(base, 0) + (size_t)l * EMB * EMB, P(base, 1) + (size_t)l * EMB, lp);
    proj_kernel<<<proj_blocks, 256, 0, stream>>>(left, N,
        P(base, 3) + (size_t)l * EMB * EMB, nullptr, rp);
    zero_kernel<<<zero_blocks, 256, 0, stream>>>(agg, N * EMB);
    edge_kernel<<<edge_blocks, 256, 0, stream>>>(lp, rp, irow, jrow, ef_raw,
        e_shift, e_scale,
        P(base, 2) + (size_t)l * EMB,        // W_edge[l]
        P(base, 4) + l,                      // s_fin[l]
        P(base, 5) + (size_t)l * EMB * EMB,  // W_fin[l]
        P(base, 6) + (size_t)l * EMB,        // b_fin[l]
        agg, E);
    out_kernel<<<out_blocks, 128, 0, stream>>>(agg, right, N,
        P(base, 7) + l,                          // s_post[l]
        P(base, 8) + (size_t)l * 2 * EMB * EMB,  // W_o1[l]
        P(base, 9) + (size_t)l * EMB,            // b_o1[l]
        P(base, 10) + (size_t)l * EMB * EMB,     // W_o2[l]
        P(base, 11) + (size_t)l * EMB,           // b_o2[l]
        dst);
  };

  for (int l = 0; l < NL; ++l) {
    // V = conv(V, rev, ef, C, params_vc): left=V right=C, i=ei0 j=ei1
    conv(V, C, V, ei0, ei1, 18, l);
    // C = conv(C, ei, ef, V, params_cv): left=C right=V, i=ei1 j=ei0
    conv(C, V, C, ei1, ei0, 30, l);
  }
}
